// QuantLayer_18648747999251
// MI455X (gfx1250) — matrix operand order repaired and checked by reading, not compiled, not run on hardware
//
#include <hip/hip_runtime.h>
#include <math.h>

typedef __attribute__((ext_vector_type(8)))  _Float16 half8;
typedef __attribute__((ext_vector_type(16))) _Float16 half16;
typedef __attribute__((ext_vector_type(8)))  float    v8f;

#define IN_F   4096
#define OUT_F  4096
#define BATCH  2048
#define NELEM  16777216ull   // OUT_F * IN_F
#define MAXPOS 6
#define MAXNEG 2

// rank window for layer-0 threshold (exact Python ints):
// n=16777216, target=int(n*0.93)=15602810, win=n//20=838860
// s = target - win//2 = 15183380 ; e = target + win//2 = 16022240
#define RANK_S  15183380u
#define RANK_E  16022240u
#define RANK_E1 16022239u
#define WIN_W   838860.0

struct Scal {
  float mean, stdv, thr;
  unsigned bin_s, bin_e, below_s, below_e;
  float v_s, v_e;
  unsigned L_s, C_s, L_e, C_e;
};
struct RowP { float kp, kn; int pc, nc; };

// workspace offsets (bytes)
#define OFF_SCAL 0u
#define OFF_PART 1024u       // 2048 doubles (stat pairs) / 1024 doubles (mid sum)
#define OFF_ROWP 32768u      // 4096 * 16B
#define OFF_HTOP 98304u      // 65536 u32
#define OFF_HS   360448u     // 65536 u32 (contiguous after HTOP)
#define OFF_HE   622592u     // 65536 u32 (contiguous after HS)
#define OFF_XH   1048576u    // 2048*4096 f16 = 16 MB
#define OFF_WQ   18874368u   // 4096*4096 f16 = 32 MB

#define NB_RED 1024

// ---------------- wave / block reduction helpers ----------------
__device__ inline float waveMaxF(float v) {
  for (int i = 16; i > 0; i >>= 1) v = fmaxf(v, __shfl_xor(v, i, 32));
  return v;
}
__device__ inline int waveSumI(int v) {
  for (int i = 16; i > 0; i >>= 1) v += __shfl_xor(v, i, 32);
  return v;
}

// ---------------- CDNA5 async global->LDS (ASYNCcnt) ----------------
// per-lane copy of 16B: global memory -> LDS; tracked by ASYNCcnt
__device__ inline void asyncCopy16(const _Float16* g, const _Float16* l) {
  unsigned lds = (unsigned)(size_t)l;   // low 32 bits = wave-relative LDS addr
  asm volatile("global_load_async_to_lds_b128 %0, %1, off"
               :: "v"(lds), "v"(g) : "memory");
}
__device__ inline void asyncWait0() {
  asm volatile("s_wait_asynccnt 0x0" ::: "memory");
}
// allow the 2 just-issued copies to remain in flight; guarantees the
// previous stage (in-order completion) has fully landed in LDS
__device__ inline void asyncWait2() {
  asm volatile("s_wait_asynccnt 0x2" ::: "memory");
}

// ---------------- global mean / std (unbiased) ----------------
__global__ __launch_bounds__(256) void reduceStatK(const float* __restrict__ w,
                                                   double* __restrict__ part) {
  __shared__ double s0[256], s1[256];
  size_t tid = (size_t)blockIdx.x * 256 + threadIdx.x;
  size_t stride = (size_t)NB_RED * 256;
  double a = 0.0, b = 0.0;
  for (size_t i = tid; i < NELEM; i += stride) {
    double x = (double)w[i];
    a += x; b += x * x;
  }
  s0[threadIdx.x] = a; s1[threadIdx.x] = b; __syncthreads();
  for (int off = 128; off > 0; off >>= 1) {
    if ((int)threadIdx.x < off) {
      s0[threadIdx.x] += s0[threadIdx.x + off];
      s1[threadIdx.x] += s1[threadIdx.x + off];
    }
    __syncthreads();
  }
  if (threadIdx.x == 0) {
    part[2 * blockIdx.x]     = s0[0];
    part[2 * blockIdx.x + 1] = s1[0];
  }
}

__global__ __launch_bounds__(256) void finalStatK(const double* __restrict__ part,
                                                  Scal* __restrict__ sc) {
  __shared__ double s0[256], s1[256];
  double a = 0.0, b = 0.0;
  for (int i = threadIdx.x; i < NB_RED; i += 256) {
    a += part[2 * i]; b += part[2 * i + 1];
  }
  s0[threadIdx.x] = a; s1[threadIdx.x] = b; __syncthreads();
  for (int off = 128; off > 0; off >>= 1) {
    if ((int)threadIdx.x < off) {
      s0[threadIdx.x] += s0[threadIdx.x + off];
      s1[threadIdx.x] += s1[threadIdx.x + off];
    }
    __syncthreads();
  }
  if (threadIdx.x == 0) {
    double n = (double)NELEM;
    double mean = s0[0] / n;
    double var  = (s1[0] - s0[0] * s0[0] / n) / (n - 1.0);
    sc->mean = (float)mean;
    sc->stdv = (float)sqrt(var > 0.0 ? var : 0.0);
  }
}

// ---------------- per-row kth values (duplicate-correct top_k) ----------------
__global__ __launch_bounds__(32) void rowStatsK(const float* __restrict__ w,
                                                const float* __restrict__ mask,
                                                const Scal* __restrict__ sc,
                                                RowP* __restrict__ rp) {
  __shared__ float sv[IN_F];
  const int row  = blockIdx.x;
  const int lane = threadIdx.x;
  const float mean = sc->mean;
  const float inv  = 1.0f / (sc->stdv + 1e-8f);
  const size_t base = (size_t)row * IN_F;
  int pc = 0, nc = 0;
  for (int c = lane; c < IN_F; c += 32) {
    float v = (w[base + c] - mean) * inv;
    v = fminf(fmaxf(v, -1.0f), 1.0f);
    float valid = v * mask[base + c];
    sv[c] = valid;
    pc += (valid > 0.0f);
    nc += (valid < 0.0f);
  }
  __syncthreads();
  pc = waveSumI(pc);
  nc = waveSumI(nc);

  float kp = -__builtin_inff();
  if (pc > MAXPOS) {
    float bound = __builtin_inff();
    int rem = MAXPOS;
    for (int it = 0; it < MAXPOS; ++it) {
      float lm = -__builtin_inff();
      for (int c = lane; c < IN_F; c += 32) {
        float t = sv[c];
        if (t > 0.0f && t < bound) lm = fmaxf(lm, t);
      }
      float m = waveMaxF(lm);
      if (!(m > 0.0f)) { kp = m; break; }
      int lc = 0;
      for (int c = lane; c < IN_F; c += 32) lc += (sv[c] == m);
      int cnt = waveSumI(lc);
      if (cnt >= rem) { kp = m; break; }
      rem -= cnt; bound = m;
    }
  }
  float kn = -__builtin_inff();
  if (nc > MAXNEG) {
    float bound = __builtin_inff();
    int rem = MAXNEG;
    for (int it = 0; it < MAXNEG; ++it) {
      float lm = -__builtin_inff();
      for (int c = lane; c < IN_F; c += 32) {
        float t = sv[c];
        if (t < 0.0f) { float g = -t; if (g < bound) lm = fmaxf(lm, g); }
      }
      float m = waveMaxF(lm);
      if (!(m > 0.0f)) { kn = m; break; }
      int lc = 0;
      for (int c = lane; c < IN_F; c += 32) lc += (sv[c] < 0.0f && -sv[c] == m);
      int cnt = waveSumI(lc);
      if (cnt >= rem) { kn = m; break; }
      rem -= cnt; bound = m;
    }
  }
  if (lane == 0) { rp[row].kp = kp; rp[row].kn = kn; rp[row].pc = pc; rp[row].nc = nc; }
}

// ---------------- recompute constrained weight element ----------------
__device__ inline float compute_wm(float wv, float mk, float mean, float inv, RowP r) {
  float v = fminf(fmaxf((wv - mean) * inv, -1.0f), 1.0f);
  float valid = v * mk;
  bool pos = valid > 0.0f;
  bool neg = valid < 0.0f;
  float w1 = (r.pc > MAXPOS) ? (((pos && valid >= r.kp) || neg) ? valid : 0.0f) : v;
  float w2;
  if (r.nc > MAXNEG) {
    float t = neg ? ((-valid >= r.kn) ? valid : 0.0f) : w1;
    w2 = t * mk;
  } else {
    w2 = w1;
  }
  w2 = fminf(fmaxf(w2, -1.0f), 1.0f);
  return w2 * mk;
}

// ---------------- histogram / exact rank selection ----------------
__global__ __launch_bounds__(256) void zeroHistK(unsigned* __restrict__ h) {
  unsigned i = blockIdx.x * 256 + threadIdx.x;
  if (i < 196608u) h[i] = 0u;
}

__global__ __launch_bounds__(256) void histTopK(const float* __restrict__ w,
                                                const float* __restrict__ mask,
                                                const Scal* __restrict__ sc,
                                                const RowP* __restrict__ rp,
                                                unsigned* __restrict__ hist) {
  const float mean = sc->mean;
  const float inv  = 1.0f / (sc->stdv + 1e-8f);
  size_t stride = (size_t)gridDim.x * blockDim.x;
  for (size_t i = (size_t)blockIdx.x * blockDim.x + threadIdx.x; i < NELEM; i += stride) {
    RowP r = rp[i >> 12];
    float wm = compute_wm(w[i], mask[i], mean, inv, r);
    unsigned bits = __float_as_uint(fabsf(wm));
    atomicAdd(&hist[bits >> 16], 1u);
  }
}

__global__ __launch_bounds__(1024) void scanTopK(const unsigned* __restrict__ hist,
                                                 Scal* __restrict__ sc) {
  __shared__ unsigned part[1024];
  __shared__ unsigned pref[1024];
  const int t = threadIdx.x;
  unsigned ssum = 0;
  for (int j = 0; j < 64; ++j) ssum += hist[t * 64 + j];
  part[t] = ssum; __syncthreads();
  if (t == 0) {
    unsigned run = 0;
    for (int i = 0; i < 1024; ++i) { pref[i] = run; run += part[i]; }
  }
  __syncthreads();
  unsigned run = pref[t];
  for (int j = 0; j < 64; ++j) {
    unsigned bin = (unsigned)t * 64 + j;
    unsigned c = hist[bin];
    if (RANK_S  >= run && RANK_S  < run + c) { sc->bin_s = bin; sc->below_s = run; }
    if (RANK_E1 >= run && RANK_E1 < run + c) { sc->bin_e = bin; sc->below_e = run; }
    run += c;
  }
}

__global__ __launch_bounds__(256) void histLoK(const float* __restrict__ w,
                                               const float* __restrict__ mask,
                                               const Scal* __restrict__ sc,
                                               const RowP* __restrict__ rp,
                                               unsigned* __restrict__ histS,
                                               unsigned* __restrict__ histE) {
  const float mean = sc->mean;
  const float inv  = 1.0f / (sc->stdv + 1e-8f);
  const unsigned bs = sc->bin_s, be = sc->bin_e;
  size_t stride = (size_t)gridDim.x * blockDim.x;
  for (size_t i = (size_t)blockIdx.x * blockDim.x + threadIdx.x; i < NELEM; i += stride) {
    RowP r = rp[i >> 12];
    float wm = compute_wm(w[i], mask[i], mean, inv, r);
    unsigned bits = __float_as_uint(fabsf(wm));
    unsigned top = bits >> 16, low = bits & 0xFFFFu;
    if (top == bs) atomicAdd(&histS[low], 1u);
    if (top == be) atomicAdd(&histE[low], 1u);
  }
}

__global__ __launch_bounds__(1024) void scanLoK(const unsigned* __restrict__ hist,
                                                Scal* __restrict__ sc, int which) {
  __shared__ unsigned part[1024];
  __shared__ unsigned pref[1024];
  const int t = threadIdx.x;
  const unsigned below  = which ? sc->below_e : sc->below_s;
  const unsigned topbin = which ? sc->bin_e   : sc->bin_s;
  const unsigned rank   = (which ? RANK_E1 : RANK_S) - below;
  unsigned ssum = 0;
  for (int j = 0; j < 64; ++j) ssum += hist[t * 64 + j];
  part[t] = ssum; __syncthreads();
  if (t == 0) {
    unsigned run = 0;
    for (int i = 0; i < 1024; ++i) { pref[i] = run; run += part[i]; }
  }
  __syncthreads();
  unsigned run = pref[t];
  for (int j = 0; j < 64; ++j) {
    unsigned low = (unsigned)t * 64 + j;
    unsigned c = hist[low];
    if (rank >= run && rank < run + c) {
      float v = __uint_as_float((topbin << 16) | low);
      if (which) { sc->v_e = v; sc->L_e = below + run; sc->C_e = c; }
      else       { sc->v_s = v; sc->L_s = below + run; sc->C_s = c; }
    }
    run += c;
  }
}

__global__ __launch_bounds__(256) void sumMidK(const float* __restrict__ w,
                                               const float* __restrict__ mask,
                                               const Scal* __restrict__ sc,
                                               const RowP* __restrict__ rp,
                                               double* __restrict__ part) {
  __shared__ double sm[256];
  const float mean = sc->mean;
  const float inv  = 1.0f / (sc->stdv + 1e-8f);
  const float vs = sc->v_s, ve = sc->v_e;
  double acc = 0.0;
  size_t stride = (size_t)gridDim.x * blockDim.x;
  for (size_t i = (size_t)blockIdx.x * blockDim.x + threadIdx.x; i < NELEM; i += stride) {
    RowP r = rp[i >> 12];
    float x = fabsf(compute_wm(w[i], mask[i], mean, inv, r));
    if (x > vs && x < ve) acc += (double)x;
  }
  sm[threadIdx.x] = acc; __syncthreads();
  for (int off = 128; off > 0; off >>= 1) {
    if ((int)threadIdx.x < off) sm[threadIdx.x] += sm[threadIdx.x + off];
    __syncthreads();
  }
  if (threadIdx.x == 0) part[blockIdx.x] = sm[0];
}

__global__ __launch_bounds__(256) void finalizeK(const double* __restrict__ part,
                                                 Scal* __restrict__ sc) {
  __shared__ double sm[256];
  double a = 0.0;
  for (int i = threadIdx.x; i < NB_RED; i += 256) a += part[i];
  sm[threadIdx.x] = a; __syncthreads();
  for (int off = 128; off > 0; off >>= 1) {
    if ((int)threadIdx.x < off) sm[threadIdx.x] += sm[threadIdx.x + off];
    __syncthreads();
  }
  if (threadIdx.x == 0) {
    float vs = sc->v_s, ve = sc->v_e;
    float thr;
    if (vs == ve) {
      thr = vs;
    } else {
      long long Ls = sc->L_s, Cs = sc->C_s, Le = sc->L_e, Ce = sc->C_e;
      long long S = (long long)RANK_S, E = (long long)RANK_E;
      long long hiS = Ls + Cs < E ? Ls + Cs : E;
      long long loS = Ls > S ? Ls : S;
      long long ts = hiS - loS; if (ts < 0) ts = 0;
      long long hiE = Le + Ce < E ? Le + Ce : E;
      long long loE = Le > S ? Le : S;
      long long te = hiE - loE; if (te < 0) te = 0;
      thr = (float)((sm[0] + (double)ts * (double)vs + (double)te * (double)ve) / WIN_W);
    }
    sc->thr = thr;
  }
}

// ---------------- quantize + convert ----------------
__global__ __launch_bounds__(256) void quantK(const float* __restrict__ w,
                                              const float* __restrict__ mask,
                                              const Scal* __restrict__ sc,
                                              const RowP* __restrict__ rp,
                                              _Float16* __restrict__ wq) {
  const float mean = sc->mean;
  const float inv  = 1.0f / (sc->stdv + 1e-8f);
  const float thr  = sc->thr;
  size_t stride = (size_t)gridDim.x * blockDim.x;
  for (size_t i = (size_t)blockIdx.x * blockDim.x + threadIdx.x; i < NELEM; i += stride) {
    RowP r = rp[i >> 12];
    float wm = compute_wm(w[i], mask[i], mean, inv, r);
    float q = (wm > thr) ? 1.0f : ((wm < -thr) ? -1.0f : 0.0f);
    wq[i] = (_Float16)q;
  }
}

__global__ __launch_bounds__(256) void convXK(const float* __restrict__ x,
                                              _Float16* __restrict__ xh) {
  size_t n = (size_t)BATCH * IN_F;
  size_t stride = (size_t)gridDim.x * blockDim.x;
  for (size_t i = (size_t)blockIdx.x * blockDim.x + threadIdx.x; i < n; i += stride)
    xh[i] = (_Float16)x[i];
}

// ---- WMMA GEMM: async global->LDS, triple-buffered, partial-count waits ----
// block = 512 threads (16 waves), tile 128(M) x 128(N), BK = 32 halves.
// Each wave computes 32x32 (2x2 WMMA accumulators). LDS rows padded to 80B so
// the 16-lane row-strided fragment reads hit 16 distinct banks.
// Pipeline: stage k+2 (2 async instrs/wave) -> compute k from LDS ->
// s_wait_asynccnt 2 (stage k+1 landed, k+2 still in flight) -> barrier.
#define BM 128
#define BN 128
#define BK 32
#define PAD_H 40   // halves per padded LDS row (80 B, 16B-aligned)

__global__ __launch_bounds__(512) void gemmK(const _Float16* __restrict__ xh,
                                             const _Float16* __restrict__ wq,
                                             float* __restrict__ out) {
  __shared__ _Float16 sA[3][BM * PAD_H];
  __shared__ _Float16 sB[3][BN * PAD_H];

  const int tid  = threadIdx.x;
  const int lane = tid & 31;
  const int wave = tid >> 5;
  const int half = lane >> 4;     // K-half selector per ISA A/B layouts
  const int l15  = lane & 15;
  const int msub = (wave >> 2) * 32;
  const int nsub = (wave & 3) * 32;
  const int tileM = blockIdx.y * BM;
  const int tileN = blockIdx.x * BN;

  // staging: each thread copies one 16B chunk of A and one of B per K-step
  const int srow = tid >> 2;            // 0..127
  const int scol = (tid & 3) * 8;       // half offset 0,8,16,24
  const _Float16* gA = xh + (size_t)(tileM + srow) * IN_F + scol;
  const _Float16* gB = wq + (size_t)(tileN + srow) * IN_F + scol;
  const int lOff = srow * PAD_H + scol; // halves within tile

  v8f c00 = {}, c01 = {}, c10 = {}, c11 = {};

  // prologue: stage k0 -> buf0, k1 -> buf1
  asyncCopy16(gA,      &sA[0][lOff]);
  asyncCopy16(gB,      &sB[0][lOff]);
  asyncCopy16(gA + BK, &sA[1][lOff]);
  asyncCopy16(gB + BK, &sB[1][lOff]);
  asyncWait2();        // buf0 landed (buf1 may still be in flight)
  __syncthreads();

  int buf = 0;
  for (int kk = 0; kk < IN_F; kk += BK) {
    const int pre = kk + 2 * BK;
    int nb = buf + 1; if (nb == 3) nb = 0;   // holds kk+BK (already staged)
    int pb = nb + 1;  if (pb == 3) pb = 0;   // re-stage target (read done @ kk-BK)
    const bool doPre = (pre < IN_F);
    if (doPre) {
      asyncCopy16(gA + pre, &sA[pb][lOff]);
      asyncCopy16(gB + pre, &sB[pb][lOff]);
    }

    const _Float16* A = &sA[buf][0];
    const _Float16* B = &sB[buf][0];
    // A frags: lane<16 holds K {0..7,16..23}; lane>=16 holds K {8..15,24..31}
    const int aoff0 = (msub + l15) * PAD_H + half * 8;
    const int aoff1 = aoff0 + 16 * PAD_H;
    half8 a0lo = *(const half8*)(A + aoff0);
    half8 a0hi = *(const half8*)(A + aoff0 + 16);
    half8 a1lo = *(const half8*)(A + aoff1);
    half8 a1hi = *(const half8*)(A + aoff1 + 16);
    half16 fa0 = __builtin_shufflevector(a0lo, a0hi, 0,1,2,3,4,5,6,7,8,9,10,11,12,13,14,15);
    half16 fa1 = __builtin_shufflevector(a1lo, a1hi, 0,1,2,3,4,5,6,7,8,9,10,11,12,13,14,15);
    // B frags: lane<16 holds K {0..7,16..23}; lane>=16 holds K {8..15,24..31} of column n (= wq row n)
    const int boff0 = (nsub + l15) * PAD_H + half * 8;
    const int boff1 = boff0 + 16 * PAD_H;
    half8 b0lo = *(const half8*)(B + boff0);
    half8 b0hi = *(const half8*)(B + boff0 + 16);
    half8 b1lo = *(const half8*)(B + boff1);
    half8 b1hi = *(const half8*)(B + boff1 + 16);
    half16 fb0 = __builtin_shufflevector(b0lo, b0hi, 0,1,2,3,4,5,6,7,8,9,10,11,12,13,14,15);
    half16 fb1 = __builtin_shufflevector(b1lo, b1hi, 0,1,2,3,4,5,6,7,8,9,10,11,12,13,14,15);

    c00 = __builtin_amdgcn_wmma_f32_16x16x32_f16(false, fa0, false, fb0, (short)0, c00, false, false);
    c01 = __builtin_amdgcn_wmma_f32_16x16x32_f16(false, fa0, false, fb1, (short)0, c01, false, false);
    c10 = __builtin_amdgcn_wmma_f32_16x16x32_f16(false, fa1, false, fb0, (short)0, c10, false, false);
    c11 = __builtin_amdgcn_wmma_f32_16x16x32_f16(false, fa1, false, fb1, (short)0, c11, false, false);

    if (doPre) asyncWait2();   // kk+BK landed; kk+2BK may still be in flight
    else       asyncWait0();   // tail: ensure final staged tile landed
    __syncthreads();           // all waves done reading buf; next buf visible
    buf = nb;
  }

  // C/D layout: VGPR r -> M = r + (lane>=16 ? 8 : 0), N = lane&15
  #pragma unroll
  for (int r = 0; r < 8; ++r) {
    int m0 = tileM + msub + half * 8 + r;
    size_t ro0 = (size_t)m0 * OUT_F;
    size_t ro1 = (size_t)(m0 + 16) * OUT_F;
    out[ro0 + tileN + nsub +  0 + l15] = c00[r];
    out[ro0 + tileN + nsub + 16 + l15] = c01[r];
    out[ro1 + tileN + nsub +  0 + l15] = c10[r];
    out[ro1 + tileN + nsub + 16 + l15] = c11[r];
  }
}

// ---------------- launcher ----------------
extern "C" void kernel_launch(void* const* d_in, const int* in_sizes, int n_in,
                              void* d_out, int out_size, void* d_ws, size_t ws_size,
                              hipStream_t stream) {
  const float* x    = (const float*)d_in[0];
  const float* w    = (const float*)d_in[1];
  const float* mask = (const float*)d_in[2];
  float* out = (float*)d_out;
  char* ws = (char*)d_ws;

  Scal*     sc   = (Scal*)(ws + OFF_SCAL);
  double*   part = (double*)(ws + OFF_PART);
  RowP*     rp   = (RowP*)(ws + OFF_ROWP);
  unsigned* hTop = (unsigned*)(ws + OFF_HTOP);
  unsigned* hS   = (unsigned*)(ws + OFF_HS);
  unsigned* hE   = (unsigned*)(ws + OFF_HE);
  _Float16* xh   = (_Float16*)(ws + OFF_XH);
  _Float16* wqd  = (_Float16*)(ws + OFF_WQ);

  zeroHistK  <<<768, 256, 0, stream>>>(hTop);            // zeros hTop+hS+hE (contiguous)
  reduceStatK<<<NB_RED, 256, 0, stream>>>(w, part);
  finalStatK <<<1, 256, 0, stream>>>(part, sc);
  rowStatsK  <<<OUT_F, 32, 0, stream>>>(w, mask, sc, rp);
  histTopK   <<<2048, 256, 0, stream>>>(w, mask, sc, rp, hTop);
  scanTopK   <<<1, 1024, 0, stream>>>(hTop, sc);
  histLoK    <<<2048, 256, 0, stream>>>(w, mask, sc, rp, hS, hE);
  scanLoK    <<<1, 1024, 0, stream>>>(hS, sc, 0);
  scanLoK    <<<1, 1024, 0, stream>>>(hE, sc, 1);
  sumMidK    <<<NB_RED, 256, 0, stream>>>(w, mask, sc, rp, part);
  finalizeK  <<<1, 256, 0, stream>>>(part, sc);
  quantK     <<<2048, 256, 0, stream>>>(w, mask, sc, rp, wqd);
  convXK     <<<2048, 256, 0, stream>>>(x, xh);

  dim3 g(OUT_F / BN, BATCH / BM);
  gemmK<<<g, 512, 0, stream>>>(xh, wqd, out);
}
